// EncoderBlock_584115552348
// MI455X (gfx1250) — compile-verified, matrix operand order checked by reading, not run on hardware
//
#include <hip/hip_runtime.h>
#include <hip/hip_bf16.h>

typedef __bf16 bf16;
typedef bf16  v16bf __attribute__((ext_vector_type(16)));
typedef bf16  v8bf  __attribute__((ext_vector_type(8)));
typedef float v8f   __attribute__((ext_vector_type(8)));
typedef int   v2i   __attribute__((ext_vector_type(2)));

typedef __attribute__((address_space(1))) void as1_void;
typedef __attribute__((address_space(3))) void as3_void;
typedef __attribute__((address_space(1))) v2i  as1_v2i;
typedef __attribute__((address_space(3))) v2i  as3_v2i;

#define D_MODEL 512
#define NHEADS  8
#define DK      64
#define D_FF    2048
#define BATCH   4
#define SEQ     2048
#define TOKENS  (BATCH * SEQ)

// ---------------------------------------------------------------------------
// CDNA5 async global->LDS copy (8 bytes per lane) + async-counter waits.
// Builtin prototype on this toolchain: (v2i addrspace(1)*, v2i addrspace(3)*,
// imm offset, imm cpol).
// ---------------------------------------------------------------------------
__device__ __forceinline__ void async_copy_b64(const bf16* gsrc, bf16* lds_dst) {
#if __has_builtin(__builtin_amdgcn_global_load_async_to_lds_b64)
    __builtin_amdgcn_global_load_async_to_lds_b64(
        (as1_v2i*)(as1_void*)gsrc,
        (as3_v2i*)(as3_void*)lds_dst, 0, 0);
#else
    unsigned loff = (unsigned)(size_t)(as3_void*)lds_dst;
    asm volatile("global_load_async_to_lds_b64 %0, %1, off"
                 :: "v"(loff), "v"(gsrc) : "memory");
#endif
}

__device__ __forceinline__ void wait_async0() {
#if __has_builtin(__builtin_amdgcn_s_wait_asynccnt)
    __builtin_amdgcn_s_wait_asynccnt(0);
#else
    asm volatile("s_wait_asynccnt 0x0" ::: "memory");
#endif
}

__device__ __forceinline__ void wait_async1() {
#if __has_builtin(__builtin_amdgcn_s_wait_asynccnt)
    __builtin_amdgcn_s_wait_asynccnt(1);
#else
    asm volatile("s_wait_asynccnt 0x1" ::: "memory");
#endif
}

// ---------------------------------------------------------------------------
// Fragment load: row-major bf16 source, 16 rows x 32 K-elements.
// CDNA5 16-bit A layout: lanes 0-15 -> row, elems 0..7 = K[h*8 .. h*8+7],
// elems 8..15 = K[16+h*8 .. +7] with h = lane>>4. B^T symmetric (lane = N).
// ---------------------------------------------------------------------------
__device__ __forceinline__ v16bf load_frag(const bf16* __restrict__ base,
                                           int stride, int row0, int k0) {
    int lane = threadIdx.x & 31;
    int r = lane & 15, h = lane >> 4;
    const bf16* p = base + (size_t)(row0 + r) * stride + (k0 + h * 8);
    v8bf lo = *(const v8bf*)(p);
    v8bf hi = *(const v8bf*)(p + 16);
    return __builtin_shufflevector(lo, hi, 0,1,2,3,4,5,6,7,8,9,10,11,12,13,14,15);
}

__device__ __forceinline__ v8f wmma_bf16(v16bf a, v16bf b, v8f c) {
    return __builtin_amdgcn_wmma_f32_16x16x32_bf16(false, a, false, b,
                                                   (short)0, c, false, false);
}

// ---------------------------------------------------------------------------
// Prep kernels
// ---------------------------------------------------------------------------
__global__ void f32_to_bf16_kernel(const float* __restrict__ in,
                                   bf16* __restrict__ out, int n) {
    int i = blockIdx.x * blockDim.x + threadIdx.x;
    if (i < n) out[i] = (bf16)in[i];
}

// w is [K,N] row-major f32 -> wT is [N,K] row-major bf16
__global__ void transpose_to_bf16_kernel(const float* __restrict__ w,
                                         bf16* __restrict__ wT, int K, int N) {
    int i = blockIdx.x * blockDim.x + threadIdx.x;
    if (i < K * N) {
        int k = i / N, n = i % N;
        wT[(size_t)n * K + k] = (bf16)w[i];
    }
}

// ---------------------------------------------------------------------------
// WMMA GEMM with async-LDS A staging.
// Workgroup: 256 threads (8 waves). Workgroup tile: 32 (M) x 512 (N);
// each wave owns a 32x64 slice (8 accumulators). The 32x32 A tile per K-step
// is DMA'd to LDS with GLOBAL_LOAD_ASYNC_TO_LDS_B64, double-buffered and
// issued one K-step ahead, shared by all 8 waves.
// mode: 0 = bf16 scatter [b,h,s,dk] (Q/K);  1 = bf16 scatter [b,h,dk,s] (V^T)
//       2 = f32 [M,N] + residual;           3 = bf16 [M,N] with ReLU
// ---------------------------------------------------------------------------
__global__ void gemm_wmma_kernel(const bf16* __restrict__ A,
                                 const bf16* __restrict__ Bt,
                                 const float* __restrict__ bias,
                                 const float* __restrict__ resid,
                                 void* __restrict__ outp,
                                 int M, int N, int K, int mode) {
    __shared__ bf16 tileA[2][32][32];

    int tid   = threadIdx.x;
    int wave  = tid >> 5;
    int gridN = N >> 9;                         // N / 512
    int tileM  = (blockIdx.x / gridN) << 5;
    int tileN0 = (blockIdx.x % gridN) << 9;
    int tN     = tileN0 + wave * 64;

    // async stage: each thread copies 8B; 256 threads cover 32x32 bf16.
    int arow = tid >> 3;                        // 0..31
    int acol = (tid & 7) * 4;                   // element offset, 8B granules
    const bf16* aSrcRow = A + (size_t)(tileM + arow) * K + acol;

    v8f acc[2][4];
#pragma unroll
    for (int mi = 0; mi < 2; ++mi)
#pragma unroll
        for (int t = 0; t < 4; ++t)
#pragma unroll
            for (int i = 0; i < 8; ++i) acc[mi][t][i] = 0.0f;

    int ksteps = K >> 5;
    async_copy_b64(aSrcRow, &tileA[0][arow][acol]);          // prime stage 0

    for (int ks = 0; ks < ksteps; ++ks) {
        int cur = ks & 1;
        if (ks + 1 < ksteps) {
            async_copy_b64(aSrcRow + (ks + 1) * 32, &tileA[cur ^ 1][arow][acol]);
            wait_async1();                                   // stage ks landed
        } else {
            wait_async0();
        }
        __syncthreads();

        const bf16* at = &tileA[cur][0][0];
        v16bf a0 = load_frag(at, 32, 0, 0);
        v16bf a1 = load_frag(at, 32, 16, 0);
        int k0 = ks * 32;
#pragma unroll
        for (int t = 0; t < 4; ++t) {
            v16bf b = load_frag(Bt, K, tN + 16 * t, k0);
            acc[0][t] = wmma_bf16(a0, b, acc[0][t]);
            acc[1][t] = wmma_bf16(a1, b, acc[1][t]);
        }
        __syncthreads();                        // readers done before re-fill
    }

    int lane = tid & 31;
    int r = lane & 15, hf = lane >> 4;
#pragma unroll
    for (int mi = 0; mi < 2; ++mi) {
#pragma unroll
        for (int t = 0; t < 4; ++t) {
#pragma unroll
            for (int i = 0; i < 8; ++i) {
                int m = tileM + mi * 16 + i + 8 * hf;   // token row
                int n = tN + t * 16 + r;                // output column
                float v = acc[mi][t][i] + bias[n];
                if (mode == 0) {                        // [b, h, s, dk]
                    int bb = m >> 11, s = m & (SEQ - 1);
                    int hd = n >> 6,  c = n & (DK - 1);
                    ((bf16*)outp)[((((size_t)bb * NHEADS + hd) * SEQ) + s) * DK + c] = (bf16)v;
                } else if (mode == 1) {                 // [b, h, dk, s]
                    int bb = m >> 11, s = m & (SEQ - 1);
                    int hd = n >> 6,  c = n & (DK - 1);
                    ((bf16*)outp)[((((size_t)bb * NHEADS + hd) * DK) + c) * SEQ + s] = (bf16)v;
                } else if (mode == 2) {                 // f32 + residual
                    size_t idx = (size_t)m * N + n;
                    ((float*)outp)[idx] = v + resid[idx];
                } else {                                // bf16 ReLU
                    ((bf16*)outp)[(size_t)m * N + n] = (bf16)fmaxf(v, 0.0f);
                }
            }
        }
    }
}

// ---------------------------------------------------------------------------
// Flash attention: one wave per 16 query rows of one (b,h). Online softmax,
// 32 keys per step; P tile C-layout -> A-layout via per-wave LDS staging.
// ---------------------------------------------------------------------------
__global__ void attn_wmma_kernel(const bf16* __restrict__ q,
                                 const bf16* __restrict__ k,
                                 const bf16* __restrict__ vT,
                                 bf16* __restrict__ attn_c) {
    __shared__ bf16 pbuf[4][16 * 32];
    int wid  = threadIdx.x >> 5;
    int lane = threadIdx.x & 31;
    int r = lane & 15, hf = lane >> 4;

    int qblocks = SEQ / 64;
    int bh = blockIdx.x / qblocks;
    int qt = (blockIdx.x % qblocks) * 64 + wid * 16;

    const bf16* qh = q  + (size_t)bh * SEQ * DK;
    const bf16* kh = k  + (size_t)bh * SEQ * DK;
    const bf16* vh = vT + (size_t)bh * DK * SEQ;

    v16bf qa0 = load_frag(qh, DK, qt, 0);
    v16bf qa1 = load_frag(qh, DK, qt, 32);

    v8f o[4];
    float mrow[8], lrow[8];
#pragma unroll
    for (int t = 0; t < 4; ++t)
#pragma unroll
        for (int i = 0; i < 8; ++i) o[t][i] = 0.0f;
#pragma unroll
    for (int i = 0; i < 8; ++i) { mrow[i] = -1e30f; lrow[i] = 0.0f; }

    for (int j = 0; j < SEQ; j += 32) {
        v8f s0, s1;
#pragma unroll
        for (int i = 0; i < 8; ++i) { s0[i] = 0.0f; s1[i] = 0.0f; }
        s0 = wmma_bf16(qa0, load_frag(kh, DK, j, 0),       s0);
        s0 = wmma_bf16(qa1, load_frag(kh, DK, j, 32),      s0);
        s1 = wmma_bf16(qa0, load_frag(kh, DK, j + 16, 0),  s1);
        s1 = wmma_bf16(qa1, load_frag(kh, DK, j + 16, 32), s1);

        float corr[8];
#pragma unroll
        for (int i = 0; i < 8; ++i) {
            float a0 = s0[i] * 0.125f;        // 1/sqrt(dk)
            float a1 = s1[i] * 0.125f;
            float mx = fmaxf(a0, a1);         // xor<16 stays inside the half
            mx = fmaxf(mx, __shfl_xor(mx, 1));
            mx = fmaxf(mx, __shfl_xor(mx, 2));
            mx = fmaxf(mx, __shfl_xor(mx, 4));
            mx = fmaxf(mx, __shfl_xor(mx, 8));
            float nm = fmaxf(mrow[i], mx);
            corr[i] = __expf(mrow[i] - nm);
            float p0 = __expf(a0 - nm);
            float p1 = __expf(a1 - nm);
            float rs = p0 + p1;
            rs += __shfl_xor(rs, 1);
            rs += __shfl_xor(rs, 2);
            rs += __shfl_xor(rs, 4);
            rs += __shfl_xor(rs, 8);
            lrow[i] = lrow[i] * corr[i] + rs;
            mrow[i] = nm;
            int m = i + 8 * hf;               // C layout -> LDS [m][key]
            pbuf[wid][m * 32 + r]      = (bf16)p0;
            pbuf[wid][m * 32 + 16 + r] = (bf16)p1;
        }
#pragma unroll
        for (int t = 0; t < 4; ++t)
#pragma unroll
            for (int i = 0; i < 8; ++i) o[t][i] *= corr[i];

        v16bf pa = load_frag(&pbuf[wid][0], 32, 0, 0);  // A layout (16x32)
#pragma unroll
        for (int t = 0; t < 4; ++t)
            o[t] = wmma_bf16(pa, load_frag(vh, SEQ, t * 16, j), o[t]);
    }

    int bb = bh >> 3, hd = bh & (NHEADS - 1);
#pragma unroll
    for (int t = 0; t < 4; ++t) {
#pragma unroll
        for (int i = 0; i < 8; ++i) {
            int srow = qt + i + 8 * hf;
            size_t tok = (size_t)bb * SEQ + srow;
            int col = hd * DK + t * 16 + r;
            attn_c[tok * D_MODEL + col] = (bf16)(o[t][i] / lrow[i]);
        }
    }
}

// ---------------------------------------------------------------------------
// LayerNorm (faithful: unbiased std, eps added to std). One wave per row.
// ---------------------------------------------------------------------------
__global__ void layernorm_kernel(const float* __restrict__ z,
                                 const float* __restrict__ alpha,
                                 const float* __restrict__ beta,
                                 float* __restrict__ out32,
                                 bf16* __restrict__ out16) {
    int wid  = threadIdx.x >> 5;
    int lane = threadIdx.x & 31;
    int row = blockIdx.x * (blockDim.x >> 5) + wid;
    const float* zr = z + (size_t)row * D_MODEL;

    float x[16];
    float sum = 0.0f;
#pragma unroll
    for (int e = 0; e < 16; ++e) { x[e] = zr[lane + 32 * e]; sum += x[e]; }
    sum += __shfl_xor(sum, 1);  sum += __shfl_xor(sum, 2);
    sum += __shfl_xor(sum, 4);  sum += __shfl_xor(sum, 8);
    sum += __shfl_xor(sum, 16);
    float mean = sum * (1.0f / 512.0f);

    float var = 0.0f;
#pragma unroll
    for (int e = 0; e < 16; ++e) { float d = x[e] - mean; var += d * d; }
    var += __shfl_xor(var, 1);  var += __shfl_xor(var, 2);
    var += __shfl_xor(var, 4);  var += __shfl_xor(var, 8);
    var += __shfl_xor(var, 16);
    var *= (1.0f / 511.0f);                   // ddof = 1
    float inv = 1.0f / (sqrtf(var) + 1e-6f);  // eps added to std

#pragma unroll
    for (int e = 0; e < 16; ++e) {
        int c = lane + 32 * e;
        float y = alpha[c] * (x[e] - mean) * inv + beta[c];
        out32[(size_t)row * D_MODEL + c] = y;
        if (out16) out16[(size_t)row * D_MODEL + c] = (bf16)y;
    }
}

// ---------------------------------------------------------------------------
// Launch
// ---------------------------------------------------------------------------
extern "C" void kernel_launch(void* const* d_in, const int* in_sizes, int n_in,
                              void* d_out, int out_size, void* d_ws, size_t ws_size,
                              hipStream_t stream) {
    const float* x    = (const float*)d_in[0];
    const float* wq   = (const float*)d_in[1];
    const float* bq   = (const float*)d_in[2];
    const float* wk   = (const float*)d_in[3];
    const float* bk   = (const float*)d_in[4];
    const float* wv   = (const float*)d_in[5];
    const float* bv   = (const float*)d_in[6];
    const float* wo   = (const float*)d_in[7];
    const float* bo   = (const float*)d_in[8];
    const float* w1   = (const float*)d_in[9];
    const float* b1   = (const float*)d_in[10];
    const float* w2   = (const float*)d_in[11];
    const float* b2   = (const float*)d_in[12];
    const float* ln1a = (const float*)d_in[13];
    const float* ln1b = (const float*)d_in[14];
    const float* ln2a = (const float*)d_in[15];
    const float* ln2b = (const float*)d_in[16];
    float* out = (float*)d_out;

    char* ws = (char*)d_ws;
    size_t off = 0;
    auto carve = [&](size_t bytes) { char* p = ws + off; off += (bytes + 255) & ~(size_t)255; return p; };
    bf16*  xb   = (bf16*)carve((size_t)TOKENS * D_MODEL * 2);
    bf16*  wqT  = (bf16*)carve((size_t)D_MODEL * D_MODEL * 2);
    bf16*  wkT  = (bf16*)carve((size_t)D_MODEL * D_MODEL * 2);
    bf16*  wvT  = (bf16*)carve((size_t)D_MODEL * D_MODEL * 2);
    bf16*  woT  = (bf16*)carve((size_t)D_MODEL * D_MODEL * 2);
    bf16*  w1T  = (bf16*)carve((size_t)D_MODEL * D_FF * 2);
    bf16*  w2T  = (bf16*)carve((size_t)D_FF * D_MODEL * 2);
    bf16*  qb   = (bf16*)carve((size_t)TOKENS * D_MODEL * 2);
    bf16*  kb   = (bf16*)carve((size_t)TOKENS * D_MODEL * 2);
    bf16*  vTb  = (bf16*)carve((size_t)TOKENS * D_MODEL * 2);
    bf16*  attn = (bf16*)carve((size_t)TOKENS * D_MODEL * 2);
    float* z1   = (float*)carve((size_t)TOKENS * D_MODEL * 4);   // reused for z2
    float* y32  = (float*)carve((size_t)TOKENS * D_MODEL * 4);
    bf16*  y16  = (bf16*)carve((size_t)TOKENS * D_MODEL * 2);
    bf16*  ffh  = (bf16*)carve((size_t)TOKENS * D_FF * 2);

    // --- prep: bf16 conversion + weight transposes ---
    f32_to_bf16_kernel<<<(TOKENS * D_MODEL) / 256, 256, 0, stream>>>(x, xb, TOKENS * D_MODEL);
    transpose_to_bf16_kernel<<<(D_MODEL * D_MODEL) / 256, 256, 0, stream>>>(wq, wqT, D_MODEL, D_MODEL);
    transpose_to_bf16_kernel<<<(D_MODEL * D_MODEL) / 256, 256, 0, stream>>>(wk, wkT, D_MODEL, D_MODEL);
    transpose_to_bf16_kernel<<<(D_MODEL * D_MODEL) / 256, 256, 0, stream>>>(wv, wvT, D_MODEL, D_MODEL);
    transpose_to_bf16_kernel<<<(D_MODEL * D_MODEL) / 256, 256, 0, stream>>>(wo, woT, D_MODEL, D_MODEL);
    transpose_to_bf16_kernel<<<(D_MODEL * D_FF) / 256, 256, 0, stream>>>(w1, w1T, D_MODEL, D_FF);
    transpose_to_bf16_kernel<<<(D_FF * D_MODEL) / 256, 256, 0, stream>>>(w2, w2T, D_FF, D_MODEL);

    // --- QKV projections: workgroup tile 32x512 -> 256 blocks each ---
    gemm_wmma_kernel<<<256, 256, 0, stream>>>(xb, wqT, bq, nullptr, qb,  TOKENS, D_MODEL, D_MODEL, 0);
    gemm_wmma_kernel<<<256, 256, 0, stream>>>(xb, wkT, bk, nullptr, kb,  TOKENS, D_MODEL, D_MODEL, 0);
    gemm_wmma_kernel<<<256, 256, 0, stream>>>(xb, wvT, bv, nullptr, vTb, TOKENS, D_MODEL, D_MODEL, 1);

    // --- attention: 32 (b,h) * 32 q-blocks, 4 waves each ---
    attn_wmma_kernel<<<BATCH * NHEADS * (SEQ / 64), 128, 0, stream>>>(qb, kb, vTb, attn);

    // --- out-proj + residual -> z1; LN1 -> y (f32 + bf16) ---
    gemm_wmma_kernel<<<256, 256, 0, stream>>>(attn, woT, bo, x, z1, TOKENS, D_MODEL, D_MODEL, 2);
    layernorm_kernel<<<TOKENS / 8, 256, 0, stream>>>(z1, ln1a, ln1b, y32, y16);

    // --- FFN1 (ReLU) -> ffh; FFN2 + residual(y) -> z1; LN2 -> out ---
    gemm_wmma_kernel<<<1024, 256, 0, stream>>>(y16, w1T, b1, nullptr, ffh, TOKENS, D_FF, D_MODEL, 3);
    gemm_wmma_kernel<<<256, 256, 0, stream>>>(ffh, w2T, b2, y32, z1, TOKENS, D_MODEL, D_FF, 2);
    layernorm_kernel<<<TOKENS / 8, 256, 0, stream>>>(z1, ln2a, ln2b, out, nullptr);
}
